// tract_DGCNN_cls_28132035789507
// MI455X (gfx1250) — compile-verified, matrix operand order checked by reading, not run on hardware
//
#include <hip/hip_runtime.h>
#include <hip/hip_bf16.h>
#include <math.h>

// ---------------- types ----------------
typedef __bf16 bf16_t;
typedef __attribute__((ext_vector_type(16))) __bf16 v16bf;
typedef __attribute__((ext_vector_type(8)))  __bf16 v8bf;
typedef __attribute__((ext_vector_type(4)))  __bf16 v4bf;
typedef __attribute__((ext_vector_type(8)))  float  v8f;
typedef __attribute__((ext_vector_type(4)))  float  v4f;

#define NPTS 15
#define NCLD 2048
#define MROWS (NCLD * NPTS)   // 30720
#define KNN  5
#define SLOPE 0.2f
#define EPSBN 1e-5f

// ---------------- workspace layout (bytes, all 16B-aligned) ----------------
static constexpr size_t OFF_ACT0  = 0;                                               // 30720*3 f32
static constexpr size_t OFF_IDX   = OFF_ACT0 + (size_t)MROWS * 3 * sizeof(float);    // 30720*5 i32
static constexpr size_t OFF_CAT   = OFF_IDX  + (size_t)MROWS * KNN * sizeof(int);    // 30720*512 f32
static constexpr size_t OFF_T     = OFF_CAT  + (size_t)MROWS * 512 * sizeof(float);  // 30720*1024 f32
static constexpr size_t OFF_B     = OFF_T    + (size_t)MROWS * 1024 * sizeof(float); // 1024*2048 bf16
static constexpr size_t OFF_STATS = OFF_B    + (size_t)2048 * 1024 * sizeof(bf16_t); // 4096 f32
static constexpr size_t OFF_POOL  = OFF_STATS + 4096 * sizeof(float);                // 2048*2048 f32

// ---------------- transpose x: (N,3,P) -> (N*P, 3) ----------------
__global__ void transpose_x_kernel(const float* __restrict__ x, float* __restrict__ act0) {
    int i = blockIdx.x * blockDim.x + threadIdx.x;
    if (i >= MROWS * 3) return;
    int m = i / 3, c = i % 3;
    int n = m / NPTS, p = m % NPTS;
    act0[i] = x[(size_t)n * 3 * NPTS + c * NPTS + p];
}

// ---------------- kNN (top-5 of -||xi-xj||^2 per row, ties -> lower idx) ----------------
__global__ void knn_kernel(const float* __restrict__ act, int lda, int C,
                           int* __restrict__ idx) {
    __shared__ float xl[NPTS * 128];   // max C used for kNN is 128
    int n = blockIdx.x;
    int tid = threadIdx.x;
    for (int i = tid; i < NPTS * C; i += 32) {
        int p = i / C, c = i % C;
        xl[p * C + c] = act[(size_t)(n * NPTS + p) * lda + c];
    }
    __syncthreads();
    if (tid < NPTS) {
        int p = tid;
        float pd[NPTS];
        for (int q = 0; q < NPTS; ++q) {
            float d = 0.f;
            for (int c = 0; c < C; ++c) {
                float t = xl[p * C + c] - xl[q * C + c];
                d += t * t;
            }
            pd[q] = -d;
        }
        bool used[NPTS] = {};
        for (int s = 0; s < KNN; ++s) {
            float best = -3.4e38f; int bi = 0;
            for (int q = 0; q < NPTS; ++q)
                if (!used[q] && pd[q] > best) { best = pd[q]; bi = q; }
            used[bi] = true;
            idx[((size_t)n * NPTS + p) * KNN + s] = bi;
        }
    }
}

// ---------------- weight prep ----------------
// Edge W (Cout, 2*Cin) -> Bg n-major [2*Cout][Cin] bf16:
//   rows [0,Cout)   = Wa          (o2 = o)
//   rows [Cout,2Co) = (Wb - Wa)   (o2 = Cout + o)
__global__ void prep_edge_w_kernel(const float* __restrict__ W, int Cout, int Cin,
                                   bf16_t* __restrict__ Bg) {
    int i = blockIdx.x * blockDim.x + threadIdx.x;
    if (i >= 2 * Cout * Cin) return;
    int o2 = i / Cin, c = i % Cin;
    float v;
    if (o2 < Cout) v = W[(size_t)o2 * (2 * Cin) + c];
    else { int o = o2 - Cout; v = W[(size_t)o * (2 * Cin) + Cin + c] - W[(size_t)o * (2 * Cin) + c]; }
    Bg[(size_t)o2 * Cin + c] = (bf16_t)v;
}

// FC weight (Out, In) f32 -> bf16, same (n-major) layout: pure cast
__global__ void prep_cast_kernel(const float* __restrict__ W, int count,
                                 bf16_t* __restrict__ Bg) {
    int i = blockIdx.x * blockDim.x + threadIdx.x;
    if (i >= count) return;
    Bg[i] = (bf16_t)W[i];
}

// ---------------- bf16 WMMA GEMM: C[MxN] = A[MxK](f32) * Bg (+bias) ----------------
// Bg is n-major: Bg[n*ldb + k], ldb == Kdim.
// Block tile 128x64, 8 waves (4m x 2n), each wave 32x32 = 2x2 WMMA tiles.
#define BM 128
#define BN 64
#define BK 32
#define LPAD 8   // pad LDS rows to 40 bf16 (80B): conflict-free ds_load_b128

__global__ __launch_bounds__(256)
void gemm_bf16_wmma(const float* __restrict__ A, int lda, int Kdim,
                    const bf16_t* __restrict__ Bg, int ldb,
                    float* __restrict__ C, int ldc,
                    const float* __restrict__ bias,
                    int M, int Nout) {
    __shared__ __align__(16) bf16_t As[BM][BK + LPAD];  // m-major, k contiguous
    __shared__ __align__(16) bf16_t Bs[BN][BK + LPAD];  // n-major, k contiguous
    const int tid  = threadIdx.x;
    const int lane = tid & 31;
    const int wave = tid >> 5;          // 0..7
    const int wm = wave & 3;            // 4 row-waves  (32 rows each)
    const int wn = wave >> 2;           // 2 col-waves  (32 cols each)
    const int rowBase = blockIdx.x * BM;
    const int colBase = blockIdx.y * BN;
    const int half = lane >> 4;
    const int l16  = lane & 15;

    v8f acc00 = {}; v8f acc01 = {};     // (m-tile 0) x (n-tile 0/1)
    v8f acc10 = {}; v8f acc11 = {};     // (m-tile 1) x (n-tile 0/1)

    // per-thread staging coordinates (fast path)
    const int ar = tid >> 3;            // A chunk row within tile (+32 per it)
    const int ac4 = (tid & 7) * 4;      // A chunk col (f32 quad)
    const int br = tid >> 2;            // B row (n) within tile
    const int bc8 = (tid & 3) * 8;      // B col (bf16 oct)

    const bool fullMN = (rowBase + BM <= M) && (colBase + BN <= Nout) &&
                        ((lda & 3) == 0) && ((ldb & 7) == 0);
    const int kTiles = (Kdim + BK - 1) / BK;
    for (int kt = 0; kt < kTiles; ++kt) {
        const int k0g = kt * BK;
        if (fullMN && (k0g + BK <= Kdim)) {
            // ---- fast path ----
            // Phase 1: issue ALL global loads back-to-back (clause; staggered waits)
            v4f va[4];
#pragma unroll
            for (int it = 0; it < 4; ++it) {
                va[it] = *(const v4f*)(A + (size_t)(rowBase + ar + it * 32) * lda + k0g + ac4);
            }
            v8bf vb = *(const v8bf*)(Bg + (size_t)(colBase + br) * ldb + k0g + bc8);
            // Phase 2: convert + LDS stores
#pragma unroll
            for (int it = 0; it < 4; ++it) {
                v4bf w;
                w[0] = (bf16_t)va[it][0]; w[1] = (bf16_t)va[it][1];
                w[2] = (bf16_t)va[it][2]; w[3] = (bf16_t)va[it][3];
                *(v4bf*)&As[ar + it * 32][ac4] = w;
            }
            *(v8bf*)&Bs[br][bc8] = vb;
            // prefetch next k-tile (global_prefetch_b8)
            if (k0g + BK < Kdim) {
                __builtin_prefetch(A + (size_t)(rowBase + (tid >> 1)) * lda + k0g + BK, 0, 1);
                __builtin_prefetch(Bg + (size_t)(colBase + br) * ldb + k0g + BK, 0, 1);
            }
        } else {
            // ---- guarded edge path ----
            for (int i = tid; i < BM * BK; i += 256) {
                int r = i >> 5, c = i & 31;
                int gr = rowBase + r, gc = k0g + c;
                float v = (gr < M && gc < Kdim) ? A[(size_t)gr * lda + gc] : 0.f;
                As[r][c] = (bf16_t)v;
            }
            for (int i = tid; i < BN * BK; i += 256) {
                int r = i >> 5, c = i & 31;     // r = n within tile, c = k
                int gn = colBase + r, gk = k0g + c;
                Bs[r][c] = (gn < Nout && gk < Kdim) ? Bg[(size_t)gn * ldb + gk] : (bf16_t)0.f;
            }
        }
        __syncthreads();

        // fragments: per ISA, lane(half,m) needs k in [8h,8h+8) and [16+8h,16+8h+8):
        // two contiguous 16B runs -> ds_load_b128 x2 each; 2x2 reg blocking reuses each twice
        const int am0 = wm * 32 + l16;
        v8bf a00 = *(const v8bf*)&As[am0][8 * half];
        v8bf a01 = *(const v8bf*)&As[am0][16 + 8 * half];
        v16bf afrag0 = __builtin_shufflevector(a00, a01, 0,1,2,3,4,5,6,7,8,9,10,11,12,13,14,15);
        v8bf a10 = *(const v8bf*)&As[am0 + 16][8 * half];
        v8bf a11 = *(const v8bf*)&As[am0 + 16][16 + 8 * half];
        v16bf afrag1 = __builtin_shufflevector(a10, a11, 0,1,2,3,4,5,6,7,8,9,10,11,12,13,14,15);

        const int bn0 = wn * 32 + l16;
        v8bf b00 = *(const v8bf*)&Bs[bn0][8 * half];
        v8bf b01 = *(const v8bf*)&Bs[bn0][16 + 8 * half];
        v16bf bfrag0 = __builtin_shufflevector(b00, b01, 0,1,2,3,4,5,6,7,8,9,10,11,12,13,14,15);
        v8bf b10 = *(const v8bf*)&Bs[bn0 + 16][8 * half];
        v8bf b11 = *(const v8bf*)&Bs[bn0 + 16][16 + 8 * half];
        v16bf bfrag1 = __builtin_shufflevector(b10, b11, 0,1,2,3,4,5,6,7,8,9,10,11,12,13,14,15);

        acc00 = __builtin_amdgcn_wmma_f32_16x16x32_bf16(false, afrag0, false, bfrag0,
                                                        (short)0, acc00, false, false);
        acc01 = __builtin_amdgcn_wmma_f32_16x16x32_bf16(false, afrag0, false, bfrag1,
                                                        (short)0, acc01, false, false);
        acc10 = __builtin_amdgcn_wmma_f32_16x16x32_bf16(false, afrag1, false, bfrag0,
                                                        (short)0, acc10, false, false);
        acc11 = __builtin_amdgcn_wmma_f32_16x16x32_bf16(false, afrag1, false, bfrag1,
                                                        (short)0, acc11, false, false);
        __syncthreads();
    }

    // store D (VGPR r, lane L: m = r + 8*(L>>4), n = L&15)
    const int outRow0 = rowBase + wm * 32;
    const int col0 = colBase + wn * 32 + l16;
    const int col1 = col0 + 16;
    const bool c0ok = col0 < Nout;
    const bool c1ok = col1 < Nout;
    const float bia0 = (bias && c0ok) ? bias[col0] : 0.f;
    const float bia1 = (bias && c1ok) ? bias[col1] : 0.f;
#pragma unroll
    for (int r = 0; r < 8; ++r) {
        int m0 = outRow0 + r + 8 * half;
        int m1 = m0 + 16;
        if (m0 < M) {
            if (c0ok) C[(size_t)m0 * ldc + col0] = acc00[r] + bia0;
            if (c1ok) C[(size_t)m0 * ldc + col1] = acc01[r] + bia1;
        }
        if (m1 < M) {
            if (c0ok) C[(size_t)m1 * ldc + col0] = acc10[r] + bia0;
            if (c1ok) C[(size_t)m1 * ldc + col1] = acc11[r] + bia1;
        }
    }
}

// ---------------- edge-block BN stats over (n,p,k): one block per channel ----------------
__global__ __launch_bounds__(256)
void edge_stats_kernel(const float* __restrict__ t, const int* __restrict__ idx,
                       int M, int Cout, float* __restrict__ stats) {
    const int o = blockIdx.x, tid = threadIdx.x;
    const int twoC = 2 * Cout;
    float s1 = 0.f, s2 = 0.f;
    for (int m = tid; m < M; m += 256) {
        int n = m / NPTS;
        float t2 = t[(size_t)m * twoC + Cout + o];
        const int* ip = idx + (size_t)m * KNN;
        for (int k = 0; k < KNN; ++k) {
            float y = t[(size_t)(n * NPTS + ip[k]) * twoC + o] + t2;
            s1 += y; s2 += y * y;
        }
    }
    __shared__ float r1[256], r2[256];
    r1[tid] = s1; r2[tid] = s2; __syncthreads();
    for (int off = 128; off > 0; off >>= 1) {
        if (tid < off) { r1[tid] += r1[tid + off]; r2[tid] += r2[tid + off]; }
        __syncthreads();
    }
    if (tid == 0) {
        float cnt = (float)M * (float)KNN;
        float mean = r1[0] / cnt;
        stats[o] = mean;
        stats[Cout + o] = r2[0] / cnt - mean * mean;
    }
}

// ---------------- edge finalize: gather+add, BN, LReLU, max over K -> cat column ----------------
__global__ void edge_finalize_kernel(const float* __restrict__ t, const int* __restrict__ idx,
                                     const float* __restrict__ stats,
                                     const float* __restrict__ g, const float* __restrict__ b,
                                     float* __restrict__ out, int outStride, int M, int Cout) {
    int i = blockIdx.x * blockDim.x + threadIdx.x;
    if (i >= M * Cout) return;
    int m = i / Cout, o = i % Cout;
    int n = m / NPTS;
    const int twoC = 2 * Cout;
    float mean = stats[o];
    float scale = rsqrtf(stats[Cout + o] + EPSBN) * g[o];
    float beta = b[o];
    float t2 = t[(size_t)m * twoC + Cout + o];
    const int* ip = idx + (size_t)m * KNN;
    float best = -3.4e38f;
    for (int k = 0; k < KNN; ++k) {
        float y = t[(size_t)(n * NPTS + ip[k]) * twoC + o] + t2;
        y = (y - mean) * scale + beta;
        y = (y >= 0.f) ? y : SLOPE * y;
        best = fmaxf(best, y);
    }
    out[(size_t)m * outStride + o] = best;
}

// ---------------- per-column stats (rows x cols), one block per column ----------------
__global__ __launch_bounds__(256)
void col_stats_kernel(const float* __restrict__ X, int rows, int cols,
                      float* __restrict__ stats) {
    const int o = blockIdx.x, tid = threadIdx.x;
    float s1 = 0.f, s2 = 0.f;
    for (int r = tid; r < rows; r += 256) {
        float v = X[(size_t)r * cols + o];
        s1 += v; s2 += v * v;
    }
    __shared__ float r1[256], r2[256];
    r1[tid] = s1; r2[tid] = s2; __syncthreads();
    for (int off = 128; off > 0; off >>= 1) {
        if (tid < off) { r1[tid] += r1[tid + off]; r2[tid] += r2[tid + off]; }
        __syncthreads();
    }
    if (tid == 0) {
        float mean = r1[0] / (float)rows;
        stats[o] = mean;
        stats[cols + o] = r2[0] / (float)rows - mean * mean;
    }
}

// ---------------- in-place BN + leaky ReLU ----------------
__global__ void bn_lrelu_kernel(float* __restrict__ X, const float* __restrict__ stats,
                                const float* __restrict__ g, const float* __restrict__ b,
                                int rows, int cols) {
    int i = blockIdx.x * blockDim.x + threadIdx.x;
    if (i >= rows * cols) return;
    int o = i % cols;
    float v = (X[i] - stats[o]) * rsqrtf(stats[cols + o] + EPSBN) * g[o] + b[o];
    X[i] = (v >= 0.f) ? v : SLOPE * v;
}

// ---------------- pooling: h (M x 1024) -> pooled (N x 2048) = [max_p | mean_p] ----------------
__global__ void pool_kernel(const float* __restrict__ h, float* __restrict__ pooled) {
    int i = blockIdx.x * blockDim.x + threadIdx.x;
    if (i >= NCLD * 1024) return;
    int n = i >> 10, o = i & 1023;
    float mx = -3.4e38f, sm = 0.f;
    for (int p = 0; p < NPTS; ++p) {
        float v = h[(size_t)(n * NPTS + p) * 1024 + o];
        mx = fmaxf(mx, v); sm += v;
    }
    pooled[(size_t)n * 2048 + o] = mx;
    pooled[(size_t)n * 2048 + 1024 + o] = sm / (float)NPTS;
}

// ---------------- in-place log-softmax per row ----------------
__global__ __launch_bounds__(256)
void log_softmax_kernel(float* __restrict__ z, int cols) {
    const int row = blockIdx.x, tid = threadIdx.x;
    __shared__ float red[256];
    float mx = -3.4e38f;
    for (int c = tid; c < cols; c += 256) mx = fmaxf(mx, z[(size_t)row * cols + c]);
    red[tid] = mx; __syncthreads();
    for (int off = 128; off > 0; off >>= 1) {
        if (tid < off) red[tid] = fmaxf(red[tid], red[tid + off]);
        __syncthreads();
    }
    float m = red[0]; __syncthreads();
    float s = 0.f;
    for (int c = tid; c < cols; c += 256) s += expf(z[(size_t)row * cols + c] - m);
    red[tid] = s; __syncthreads();
    for (int off = 128; off > 0; off >>= 1) {
        if (tid < off) red[tid] += red[tid + off];
        __syncthreads();
    }
    float lse = m + logf(red[0]); __syncthreads();
    for (int c = tid; c < cols; c += 256) z[(size_t)row * cols + c] -= lse;
}

// ---------------- host-side edge block driver ----------------
static void run_edge_block(const float* actIn, int lda, int Cin,
                           const float* W, const float* g, const float* b, int Cout,
                           float* cat, int colOff,
                           int* idxBuf, float* tBuf, bf16_t* Bbuf, float* statsBuf,
                           hipStream_t stream) {
    knn_kernel<<<NCLD, 32, 0, stream>>>(actIn, lda, Cin, idxBuf);
    int bw = 2 * Cout * Cin;
    prep_edge_w_kernel<<<(bw + 255) / 256, 256, 0, stream>>>(W, Cout, Cin, Bbuf);
    dim3 gg((MROWS + BM - 1) / BM, (2 * Cout + BN - 1) / BN);
    gemm_bf16_wmma<<<gg, 256, 0, stream>>>(actIn, lda, Cin, Bbuf, Cin,
                                           tBuf, 2 * Cout, nullptr, MROWS, 2 * Cout);
    edge_stats_kernel<<<Cout, 256, 0, stream>>>(tBuf, idxBuf, MROWS, Cout, statsBuf);
    edge_finalize_kernel<<<((MROWS * Cout) + 255) / 256, 256, 0, stream>>>(
        tBuf, idxBuf, statsBuf, g, b, cat + colOff, 512, MROWS, Cout);
}

extern "C" void kernel_launch(void* const* d_in, const int* in_sizes, int n_in,
                              void* d_out, int out_size, void* d_ws, size_t ws_size,
                              hipStream_t stream) {
    const float* x   = (const float*)d_in[0];
    const float* W1  = (const float*)d_in[2];
    const float* g1  = (const float*)d_in[3];
    const float* b1  = (const float*)d_in[4];
    const float* W2  = (const float*)d_in[5];
    const float* g2  = (const float*)d_in[6];
    const float* b2  = (const float*)d_in[7];
    const float* W3  = (const float*)d_in[8];
    const float* g3  = (const float*)d_in[9];
    const float* b3  = (const float*)d_in[10];
    const float* W4  = (const float*)d_in[11];
    const float* g4  = (const float*)d_in[12];
    const float* b4  = (const float*)d_in[13];
    const float* W5  = (const float*)d_in[14];
    const float* g5  = (const float*)d_in[15];
    const float* b5  = (const float*)d_in[16];
    const float* L1w = (const float*)d_in[17];
    const float* g6  = (const float*)d_in[18];
    const float* b6  = (const float*)d_in[19];
    const float* L2w = (const float*)d_in[20];
    const float* L2b = (const float*)d_in[21];
    const float* g7  = (const float*)d_in[22];
    const float* b7  = (const float*)d_in[23];
    const float* L3w = (const float*)d_in[24];
    const float* L3b = (const float*)d_in[25];

    char* ws = (char*)d_ws;
    float*  act0   = (float*)(ws + OFF_ACT0);
    int*    idxB   = (int*)  (ws + OFF_IDX);
    float*  cat    = (float*)(ws + OFF_CAT);    // (M x 512): x1|x2|x3|x4 at cols 0/64/128/256
    float*  tB     = (float*)(ws + OFF_T);      // (M x 1024) GEMM scratch / h
    bf16_t* Bw     = (bf16_t*)(ws + OFF_B);
    float*  stats  = (float*)(ws + OFF_STATS);
    float*  pooled = (float*)(ws + OFF_POOL);   // (2048 x 2048)

    // x (N,3,P) -> act0 (M x 3)
    transpose_x_kernel<<<((MROWS * 3) + 255) / 256, 256, 0, stream>>>(x, act0);

    // edge blocks
    run_edge_block(act0,      3,   3,   W1, g1, b1,  64, cat,   0, idxB, tB, Bw, stats, stream);
    run_edge_block(cat + 0,   512, 64,  W2, g2, b2,  64, cat,  64, idxB, tB, Bw, stats, stream);
    run_edge_block(cat + 64,  512, 64,  W3, g3, b3, 128, cat, 128, idxB, tB, Bw, stats, stream);
    run_edge_block(cat + 128, 512, 128, W4, g4, b4, 256, cat, 256, idxB, tB, Bw, stats, stream);

    // h = lrelu(bn(cat @ W5^T))  -> tB (M x 1024)
    prep_cast_kernel<<<((1024 * 512) + 255) / 256, 256, 0, stream>>>(W5, 1024 * 512, Bw);
    {
        dim3 gg((MROWS + BM - 1) / BM, 1024 / BN);
        gemm_bf16_wmma<<<gg, 256, 0, stream>>>(cat, 512, 512, Bw, 512, tB, 1024,
                                               nullptr, MROWS, 1024);
    }
    col_stats_kernel<<<1024, 256, 0, stream>>>(tB, MROWS, 1024, stats);
    bn_lrelu_kernel<<<((MROWS * 1024) + 255) / 256, 256, 0, stream>>>(tB, stats, g5, b5, MROWS, 1024);

    // pooled (2048 x 2048)
    pool_kernel<<<((NCLD * 1024) + 255) / 256, 256, 0, stream>>>(tB, pooled);

    // L1: z1 = lrelu(bn(pooled @ L1w^T))   (2048 x 512), reuse tB (h is dead)
    float* z1 = tB;
    prep_cast_kernel<<<((512 * 2048) + 255) / 256, 256, 0, stream>>>(L1w, 512 * 2048, Bw);
    {
        dim3 gg(NCLD / BM, 512 / BN);
        gemm_bf16_wmma<<<gg, 256, 0, stream>>>(pooled, 2048, 2048, Bw, 2048, z1, 512,
                                               nullptr, NCLD, 512);
    }
    col_stats_kernel<<<512, 256, 0, stream>>>(z1, NCLD, 512, stats);
    bn_lrelu_kernel<<<((NCLD * 512) + 255) / 256, 256, 0, stream>>>(z1, stats, g6, b6, NCLD, 512);

    // L2: z2 = lrelu(bn(z1 @ L2w^T + L2b))  (2048 x 256)
    float* z2 = tB + (size_t)2 * 1024 * 1024;   // 8 MB past tB start, clear of z1
    prep_cast_kernel<<<((256 * 512) + 255) / 256, 256, 0, stream>>>(L2w, 256 * 512, Bw);
    {
        dim3 gg(NCLD / BM, 256 / BN);
        gemm_bf16_wmma<<<gg, 256, 0, stream>>>(z1, 512, 512, Bw, 512, z2, 256,
                                               L2b, NCLD, 256);
    }
    col_stats_kernel<<<256, 256, 0, stream>>>(z2, NCLD, 256, stats);
    bn_lrelu_kernel<<<((NCLD * 256) + 255) / 256, 256, 0, stream>>>(z2, stats, g7, b7, NCLD, 256);

    // L3 + log_softmax -> d_out (2048 x 800)
    float* out = (float*)d_out;
    prep_cast_kernel<<<((800 * 256) + 255) / 256, 256, 0, stream>>>(L3w, 800 * 256, Bw);
    {
        dim3 gg(NCLD / BM, (800 + BN - 1) / BN);
        gemm_bf16_wmma<<<gg, 256, 0, stream>>>(z2, 256, 256, Bw, 256, out, 800,
                                               L3b, NCLD, 800);
    }
    log_softmax_kernel<<<NCLD, 256, 0, stream>>>(out, 800);
}
